// MoELayer_83021717832494
// MI455X (gfx1250) — matrix operand order repaired and checked by reading, not compiled, not run on hardware
//
#include <hip/hip_runtime.h>

// ---------------------------------------------------------------------------
// MoE layer for gfx1250 (MI455X): top-2-of-8 routed grouped GEMM in bf16 WMMA,
// with TDM (tensor_load_to_lds) double-buffered weight-tile staging.
// B=4, S=2048, D=1024 -> T=8192 tokens. E=8, H=2048, TOP_K=2.
// ---------------------------------------------------------------------------

#define T_TOK   8192
#define DIM     1024
#define HID     2048
#define NEXP    8
#define NSLOT   (2 * T_TOK)

// Workspace layout (bytes)
#define OFF_XB        0UL
#define OFF_W1B       16777216UL
#define OFF_W3B       50331648UL
#define OFF_W2B       83886080UL
#define OFF_HB        117440512UL
#define OFF_SLOT_TOK  184549376UL
#define OFF_SLOT_PRB  184614912UL
#define OFF_SEL       184680448UL
#define OFF_PROB      184745984UL
#define OFF_COUNTS    184811520UL
#define OFF_SEGOFF    184811584UL
#define OFF_CURSOR    184811648UL

typedef __attribute__((ext_vector_type(16))) __bf16 v16bf;
typedef __attribute__((ext_vector_type(8)))  float  v8f;
typedef __attribute__((ext_vector_type(4)))  unsigned int u32x4;
typedef __attribute__((ext_vector_type(8)))  int    i32x8;
typedef __attribute__((ext_vector_type(4)))  int    i32x4;
typedef unsigned short u16b;

#if __has_builtin(__builtin_amdgcn_tensor_load_to_lds) && \
    __has_builtin(__builtin_amdgcn_s_wait_tensorcnt)
#define HAVE_TDM 1
#endif

__device__ __forceinline__ u16b f2bf_bits(float f) {
    unsigned u = __builtin_bit_cast(unsigned, f);
    unsigned r = u + 0x7FFFu + ((u >> 16) & 1u);   // round-to-nearest-even
    return (u16b)(r >> 16);
}

#ifdef HAVE_TDM
// Issue a TDM 2D tile load: tile_rows x tile_cols of 2-byte elements from a
// row-major matrix with row stride `stride_e` (elements) into LDS.
// D# packing per CDNA5 ISA 8.3/8.4 (group0: count/lds/global/type=2;
// group1: data_size=2B, tensor dims, tile dims, dim0 stride).
// 6-arg builtin form (clang-23 / therock-10.0 toolchain).
__device__ __forceinline__ void tdm_load_2d(unsigned lds_off, const void* gaddr,
                                            unsigned tile_cols, unsigned tile_rows,
                                            unsigned stride_e) {
    unsigned long long ga = (unsigned long long)gaddr;
    u32x4 g0;
    g0[0] = 1u;                                            // count=1
    g0[1] = lds_off;                                       // lds_addr (bytes)
    g0[2] = (unsigned)(ga & 0xFFFFFFFFu);                  // global_addr lo
    g0[3] = (unsigned)((ga >> 32) & 0x01FFFFFFu) | 0x80000000u; // addr hi | type=2
    const unsigned tdim1 = 1u << 20;                       // generous (no OOB)
    i32x8 g1;
    g1[0] = (int)(1u << 16);                               // data_size=1 (2B)
    g1[1] = (int)((stride_e & 0xFFFFu) << 16);             // tensor_dim0 lo16
    g1[2] = (int)(((stride_e >> 16) & 0xFFFFu) | ((tdim1 & 0xFFFFu) << 16));
    g1[3] = (int)(((tdim1 >> 16) & 0xFFFFu) | (tile_cols << 16)); // | tile_dim0
    g1[4] = (int)(tile_rows & 0xFFFFu);                    // tile_dim1 (tile_dim2=0)
    g1[5] = (int)stride_e;                                 // tensor_dim0_stride lo32
    g1[6] = 0;
    g1[7] = 0;
    i32x4 z4 = {0, 0, 0, 0};                               // groups 2/3 unused (2D)
    i32x8 z8 = {0, 0, 0, 0, 0, 0, 0, 0};
    __builtin_amdgcn_tensor_load_to_lds(g0, g1, z4, z4, z8, 0);
}
__device__ __forceinline__ unsigned lds_addr_of(const void* p) {
    return (unsigned)(unsigned long long)p;   // low 32 bits = LDS offset
}
#endif

// --------------------------- init: zero out + counters ----------------------
__global__ __launch_bounds__(256) void moe_init_kernel(float* out, int* counts) {
    long gid = (long)blockIdx.x * blockDim.x + threadIdx.x;
    long i4 = gid * 4;
    if (i4 < (long)T_TOK * DIM) {
        float4 z = make_float4(0.f, 0.f, 0.f, 0.f);
        *reinterpret_cast<float4*>(out + i4) = z;
    }
    if (gid < NEXP) counts[gid] = 0;
}

// --------------------------- f32 -> bf16 conversion -------------------------
__global__ __launch_bounds__(256) void moe_convert_kernel(
    const float* __restrict__ x,  const float* __restrict__ w1,
    const float* __restrict__ w3, const float* __restrict__ w2,
    u16b* xb, u16b* w1b, u16b* w3b, u16b* w2b) {
    const long TD  = (long)T_TOK * DIM;
    const long EDH = (long)NEXP * DIM * HID;
    long gid = (long)blockIdx.x * blockDim.x + threadIdx.x;
    long i = gid * 4;
    if (i >= TD + 3 * EDH) return;
    const float* src; u16b* dst; long off;
    if (i < TD)               { src = x;  dst = xb;  off = i; }
    else if (i < TD + EDH)    { src = w1; dst = w1b; off = i - TD; }
    else if (i < TD + 2*EDH)  { src = w3; dst = w3b; off = i - TD - EDH; }
    else                      { src = w2; dst = w2b; off = i - TD - 2*EDH; }
    float4 v = *reinterpret_cast<const float4*>(src + off);
    u16b pk[4] = { f2bf_bits(v.x), f2bf_bits(v.y), f2bf_bits(v.z), f2bf_bits(v.w) };
    *reinterpret_cast<uint2*>(dst + off) = *reinterpret_cast<const uint2*>(pk);
}

// --------------------------- router: logits, top-2, softmax -----------------
__global__ __launch_bounds__(256) void moe_router_kernel(
    const float* __restrict__ x, const float* __restrict__ wg,
    int* sel, float* prob, int* counts) {
    int t    = blockIdx.x * 8 + (threadIdx.x >> 5);
    int lane = threadIdx.x & 31;
    float acc[NEXP];
    #pragma unroll
    for (int e = 0; e < NEXP; ++e) acc[e] = 0.f;
    const float* xr = x + (long)t * DIM + lane * 32;
    for (int i = 0; i < 32; ++i) {
        float xv = xr[i];
        const float* g = wg + ((long)(lane * 32 + i)) * NEXP;
        #pragma unroll
        for (int e = 0; e < NEXP; ++e) acc[e] += xv * g[e];
    }
    #pragma unroll
    for (int off = 16; off >= 1; off >>= 1) {
        #pragma unroll
        for (int e = 0; e < NEXP; ++e) acc[e] += __shfl_xor(acc[e], off, 32);
    }
    if (lane == 0) {
        int i0 = 0; float m0 = acc[0];
        #pragma unroll
        for (int e = 1; e < NEXP; ++e) if (acc[e] > m0) { m0 = acc[e]; i0 = e; }
        int i1 = -1; float m1 = -3.402823e38f;
        #pragma unroll
        for (int e = 0; e < NEXP; ++e) if (e != i0 && acc[e] > m1) { m1 = acc[e]; i1 = e; }
        float p0 = 1.f / (1.f + __expf(m1 - m0));
        float p1 = 1.f - p0;
        sel[2*t] = i0; sel[2*t+1] = i1;
        prob[2*t] = p0; prob[2*t+1] = p1;
        atomicAdd(&counts[i0], 1);
        atomicAdd(&counts[i1], 1);
    }
}

// --------------------------- prefix sum (E=8) -------------------------------
__global__ void moe_offsets_kernel(const int* counts, int* seg_off, int* cursor) {
    if (blockIdx.x == 0 && threadIdx.x == 0) {
        int s = 0;
        for (int e = 0; e < NEXP; ++e) { seg_off[e] = s; cursor[e] = s; s += counts[e]; }
    }
}

// --------------------------- scatter tokens into expert buckets -------------
__global__ __launch_bounds__(256) void moe_scatter_kernel(
    const int* sel, const float* prob, int* cursor, int* slot_tok, float* slot_prob) {
    int t = blockIdx.x * 256 + threadIdx.x;
    if (t >= T_TOK) return;
    #pragma unroll
    for (int k = 0; k < 2; ++k) {
        int e = sel[2*t + k];
        int s = atomicAdd(&cursor[e], 1);   // slot order is output-invariant
        slot_tok[s]  = t;
        slot_prob[s] = prob[2*t + k];
    }
}

// --------------------------- GEMM1: h = silu(x@w1) * (x@w3) -----------------
// Grid: (H/64, T/64, E). Block: 256 threads = 8 waves (4M x 2N wave grid).
// Tile: M=64 slots, N=64 of H (wave: 16x32), K stepped by 32 over D=1024.
// B tiles double-buffered; staged by TDM when available.
__global__ __launch_bounds__(256) void moe_gemm1_kernel(
    const u16b* __restrict__ xb, const u16b* __restrict__ w1b,
    const u16b* __restrict__ w3b, u16b* __restrict__ hb,
    const int* __restrict__ slot_tok, const int* __restrict__ seg_off,
    const int* __restrict__ counts) {
    int e   = blockIdx.z;
    int cnt = counts[e];
    int m0g = blockIdx.y * 64;
    if (m0g >= cnt) return;
    int n0g = blockIdx.x * 64;
    int seg = seg_off[e];

    __shared__ __align__(16) u16b sA[64][32];
    __shared__ __align__(16) u16b sB1[2][32][64];
    __shared__ __align__(16) u16b sB3[2][32][64];

    int tid  = threadIdx.x;
    int lane = tid & 31;
    int wid  = tid >> 5;
    int wm   = (wid & 3) * 16;       // wave rows
    int wn   = (wid >> 2) * 32;      // wave cols (two 16-wide subtiles)

    // A-tile gather loader: 4 threads x 8 halfs per row
    int arow = tid >> 2;
    int acol = (tid & 3) * 8;
    int srow = m0g + arow;
    long atok = (srow < cnt) ? (long)slot_tok[seg + srow] : -1;
    const u16b* arow_ptr = (atok >= 0) ? (xb + atok * DIM + acol) : nullptr;

    const u16b* b1base = w1b + (long)e * DIM * HID + n0g;
    const u16b* b3base = w3b + (long)e * DIM * HID + n0g;

#ifdef HAVE_TDM
    if (wid == 0) {   // prologue: stage k0=0 into buffer 0
        tdm_load_2d(lds_addr_of(&sB1[0][0][0]), b1base, 64, 32, HID);
        tdm_load_2d(lds_addr_of(&sB3[0][0][0]), b3base, 64, 32, HID);
    }
#else
    {   // prologue sync load into buffer 0
        int bh = tid >> 7, bt = tid & 127;
        int bkr = bt >> 2, bc = (bt & 3) * 16;
        const u16b* s = (bh ? b3base : b1base) + (long)bkr * HID + bc;
        u16b* d = bh ? &sB3[0][bkr][bc] : &sB1[0][bkr][bc];
        *reinterpret_cast<uint4*>(d)     = *reinterpret_cast<const uint4*>(s);
        *reinterpret_cast<uint4*>(d + 8) = *reinterpret_cast<const uint4*>(s + 8);
    }
#endif

    v8f acc1[2] = {}; v8f acc3[2] = {};
    int cur = 0;
    for (int k0 = 0; k0 < DIM; k0 += 32) {
        uint4 av = {0u, 0u, 0u, 0u};
        if (arow_ptr) av = *reinterpret_cast<const uint4*>(arow_ptr + k0);
        *reinterpret_cast<uint4*>(&sA[arow][acol]) = av;
#ifdef HAVE_TDM
        if (wid == 0) __builtin_amdgcn_s_wait_tensorcnt(0);  // cur buffer ready
#endif
        __syncthreads();

        int nxt = cur ^ 1;
        if (k0 + 32 < DIM) {   // stage next K tile, overlapped with compute
#ifdef HAVE_TDM
            if (wid == 0) {
                tdm_load_2d(lds_addr_of(&sB1[nxt][0][0]),
                            b1base + (long)(k0 + 32) * HID, 64, 32, HID);
                tdm_load_2d(lds_addr_of(&sB3[nxt][0][0]),
                            b3base + (long)(k0 + 32) * HID, 64, 32, HID);
            }
#else
            int bh = tid >> 7, bt = tid & 127;
            int bkr = bt >> 2, bc = (bt & 3) * 16;
            const u16b* s = (bh ? b3base : b1base) + (long)(k0 + 32 + bkr) * HID + bc;
            u16b* d = bh ? &sB3[nxt][bkr][bc] : &sB1[nxt][bkr][bc];
            *reinterpret_cast<uint4*>(d)     = *reinterpret_cast<const uint4*>(s);
            *reinterpret_cast<uint4*>(d + 8) = *reinterpret_cast<const uint4*>(s + 8);
#endif
        }

        // Fragments per ISA VGPR layouts (16B-aligned LDS reads)
        v16bf af;
        int m  = wm + (lane & 15);
        int kb = (lane < 16) ? 0 : 8;
        const u16b* pa = &sA[m][kb];
        #pragma unroll
        for (int j = 0; j < 8; ++j) af[j] = __builtin_bit_cast(__bf16, pa[j]);
        pa = &sA[m][kb + 16];
        #pragma unroll
        for (int j = 0; j < 8; ++j) af[8 + j] = __builtin_bit_cast(__bf16, pa[j]);

        int kk = (lane < 16) ? 0 : 8;
        #pragma unroll
        for (int sub = 0; sub < 2; ++sub) {
            v16bf b1f, b3f;
            const u16b* q1 = &sB1[cur][0][wn + sub * 16 + (lane & 15)];
            const u16b* q3 = &sB3[cur][0][wn + sub * 16 + (lane & 15)];
            #pragma unroll
            for (int j = 0; j < 16; ++j) {
                int kr = kk + (j & 7) + ((j >> 3) << 4);
                b1f[j] = __builtin_bit_cast(__bf16, q1[kr * 64]);
                b3f[j] = __builtin_bit_cast(__bf16, q3[kr * 64]);
            }
            acc1[sub] = __builtin_amdgcn_wmma_f32_16x16x32_bf16(
                false, af, false, b1f, (short)0, acc1[sub], false, false);
            acc3[sub] = __builtin_amdgcn_wmma_f32_16x16x32_bf16(
                false, af, false, b3f, (short)0, acc3[sub], false, false);
        }
        __syncthreads();   // protect sA (and retire cur buffer)
        cur = nxt;
    }

    // Epilogue: h = silu(a) * b -> bf16
    #pragma unroll
    for (int sub = 0; sub < 2; ++sub) {
        int n = n0g + wn + sub * 16 + (lane & 15);
        #pragma unroll
        for (int r = 0; r < 8; ++r) {
            int mrow = m0g + wm + r + ((lane < 16) ? 0 : 8);
            if (mrow < cnt) {
                float a = acc1[sub][r], b = acc3[sub][r];
                float h = (a / (1.f + __expf(-a))) * b;
                hb[(long)(seg + mrow) * HID + n] = f2bf_bits(h);
            }
        }
    }
}

// --------------------------- GEMM2: out += prob * (h @ w2) ------------------
// Grid: (D/64, T/64, E). Tile M=64, N=64, K stepped by 32 over H=2048.
__global__ __launch_bounds__(256) void moe_gemm2_kernel(
    const u16b* __restrict__ hb, const u16b* __restrict__ w2b,
    const int* __restrict__ slot_tok, const float* __restrict__ slot_prob,
    const int* __restrict__ seg_off, const int* __restrict__ counts,
    float* __restrict__ out) {
    int e   = blockIdx.z;
    int cnt = counts[e];
    int m0g = blockIdx.y * 64;
    if (m0g >= cnt) return;
    int n0g = blockIdx.x * 64;
    int seg = seg_off[e];

    __shared__ __align__(16) u16b sA[64][32];
    __shared__ __align__(16) u16b sB[2][32][64];

    int tid  = threadIdx.x;
    int lane = tid & 31;
    int wid  = tid >> 5;
    int wm   = (wid & 3) * 16;
    int wn   = (wid >> 2) * 32;

    int arow = tid >> 2;
    int acol = (tid & 3) * 8;
    int srow = m0g + arow;
    const u16b* arow_ptr = (srow < cnt)
        ? (hb + (long)(seg + srow) * HID + acol) : nullptr;

    const u16b* bbase = w2b + (long)e * HID * DIM + n0g;

#ifdef HAVE_TDM
    if (wid == 0) tdm_load_2d(lds_addr_of(&sB[0][0][0]), bbase, 64, 32, DIM);
#else
    {
        int bkr = tid >> 3, bc = (tid & 7) * 8;
        *reinterpret_cast<uint4*>(&sB[0][bkr][bc]) =
            *reinterpret_cast<const uint4*>(bbase + (long)bkr * DIM + bc);
    }
#endif

    v8f acc[2] = {};
    int cur = 0;
    for (int k0 = 0; k0 < HID; k0 += 32) {
        uint4 av = {0u, 0u, 0u, 0u};
        if (arow_ptr) av = *reinterpret_cast<const uint4*>(arow_ptr + k0);
        *reinterpret_cast<uint4*>(&sA[arow][acol]) = av;
#ifdef HAVE_TDM
        if (wid == 0) __builtin_amdgcn_s_wait_tensorcnt(0);
#endif
        __syncthreads();

        int nxt = cur ^ 1;
        if (k0 + 32 < HID) {
#ifdef HAVE_TDM
            if (wid == 0)
                tdm_load_2d(lds_addr_of(&sB[nxt][0][0]),
                            bbase + (long)(k0 + 32) * DIM, 64, 32, DIM);
#else
            int bkr = tid >> 3, bc = (tid & 7) * 8;
            *reinterpret_cast<uint4*>(&sB[nxt][bkr][bc]) =
                *reinterpret_cast<const uint4*>(bbase + (long)(k0 + 32 + bkr) * DIM + bc);
#endif
        }

        v16bf af;
        int m  = wm + (lane & 15);
        int kb = (lane < 16) ? 0 : 8;
        const u16b* pa = &sA[m][kb];
        #pragma unroll
        for (int j = 0; j < 8; ++j) af[j] = __builtin_bit_cast(__bf16, pa[j]);
        pa = &sA[m][kb + 16];
        #pragma unroll
        for (int j = 0; j < 8; ++j) af[8 + j] = __builtin_bit_cast(__bf16, pa[j]);

        int kk = (lane < 16) ? 0 : 8;
        #pragma unroll
        for (int sub = 0; sub < 2; ++sub) {
            v16bf bf;
            const u16b* q = &sB[cur][0][wn + sub * 16 + (lane & 15)];
            #pragma unroll
            for (int j = 0; j < 16; ++j) bf[j] = __builtin_bit_cast(__bf16, q[(kk + (j & 7) + ((j >> 3) << 4)) * 64]);
            acc[sub] = __builtin_amdgcn_wmma_f32_16x16x32_bf16(
                false, af, false, bf, (short)0, acc[sub], false, false);
        }
        __syncthreads();
        cur = nxt;
    }

    // Epilogue: prob-scaled scatter-accumulate (2 commutative f32 adds/elem)
    #pragma unroll
    for (int sub = 0; sub < 2; ++sub) {
        int n = n0g + wn + sub * 16 + (lane & 15);
        #pragma unroll
        for (int r = 0; r < 8; ++r) {
            int mrow = m0g + wm + r + ((lane < 16) ? 0 : 8);
            if (mrow < cnt) {
                int   tok = slot_tok[seg + mrow];
                float p   = slot_prob[seg + mrow];
                atomicAdd(&out[(long)tok * DIM + n], p * acc[sub][r]);
            }
        }
    }
}

// ---------------------------------------------------------------------------
extern "C" void kernel_launch(void* const* d_in, const int* in_sizes, int n_in,
                              void* d_out, int out_size, void* d_ws, size_t ws_size,
                              hipStream_t stream) {
    const float* x  = (const float*)d_in[0];
    const float* wg = (const float*)d_in[1];
    const float* w1 = (const float*)d_in[2];
    const float* w3 = (const float*)d_in[3];
    const float* w2 = (const float*)d_in[4];
    float* out = (float*)d_out;

    char* ws = (char*)d_ws;
    u16b*  xb        = (u16b*) (ws + OFF_XB);
    u16b*  w1b       = (u16b*) (ws + OFF_W1B);
    u16b*  w3b       = (u16b*) (ws + OFF_W3B);
    u16b*  w2b       = (u16b*) (ws + OFF_W2B);
    u16b*  hb        = (u16b*) (ws + OFF_HB);
    int*   slot_tok  = (int*)  (ws + OFF_SLOT_TOK);
    float* slot_prob = (float*)(ws + OFF_SLOT_PRB);
    int*   sel       = (int*)  (ws + OFF_SEL);
    float* prob      = (float*)(ws + OFF_PROB);
    int*   counts    = (int*)  (ws + OFF_COUNTS);
    int*   seg_off   = (int*)  (ws + OFF_SEGOFF);
    int*   cursor    = (int*)  (ws + OFF_CURSOR);

    {
        long n4 = ((long)T_TOK * DIM) / 4;
        moe_init_kernel<<<dim3((n4 + 255) / 256), dim3(256), 0, stream>>>(out, counts);
    }
    {
        long total = (long)T_TOK * DIM + 3L * NEXP * DIM * HID;
        long n4 = total / 4;
        moe_convert_kernel<<<dim3((n4 + 255) / 256), dim3(256), 0, stream>>>(
            x, w1, w3, w2, xb, w1b, w3b, w2b);
    }
    moe_router_kernel<<<dim3(T_TOK / 8), dim3(256), 0, stream>>>(x, wg, sel, prob, counts);
    moe_offsets_kernel<<<dim3(1), dim3(32), 0, stream>>>(counts, seg_off, cursor);
    moe_scatter_kernel<<<dim3(T_TOK / 256), dim3(256), 0, stream>>>(
        sel, prob, cursor, slot_tok, slot_prob);
    moe_gemm1_kernel<<<dim3(HID / 64, T_TOK / 64, NEXP), dim3(256), 0, stream>>>(
        xb, w1b, w3b, hb, slot_tok, seg_off, counts);
    moe_gemm2_kernel<<<dim3(DIM / 64, T_TOK / 64, NEXP), dim3(256), 0, stream>>>(
        hb, w2b, slot_tok, slot_prob, seg_off, counts, out);
}
